// StormGNN_53455162966687
// MI455X (gfx1250) — compile-verified, matrix operand order checked by reading, not run on hardware
//
#include <hip/hip_runtime.h>
#include <hip/hip_bf16.h>
#include <math.h>

// ---------------------------------------------------------------------------
// CDNA5 (gfx1250) GAT: edge-encoder + 3x(WMMA GEMM -> segment softmax -> scatter -> LN)
// ---------------------------------------------------------------------------

typedef __attribute__((ext_vector_type(16))) _Float16 v16h;
typedef __attribute__((ext_vector_type(8)))  _Float16 v8h;
typedef __attribute__((ext_vector_type(8)))  float    v8f;

#define D_   128
#define H_   4
#define C_   32
#define L_   3
#define NEG_SLOPE 0.2f
#define LN_EPS    1e-5f

union Frag { v16h v; v8h h[2]; };

__device__ __forceinline__ int enc_f(float f) {
    int i = __float_as_int(f);
    return (i >= 0) ? i : (i ^ 0x7FFFFFFF);   // monotone map float -> int
}
__device__ __forceinline__ float dec_f(int i) {
    return __int_as_float((i >= 0) ? i : (i ^ 0x7FFFFFFF));
}

// ---------------- small utility kernels ----------------
__global__ void zero_f_kernel(float* p, int n) {
    int t = blockIdx.x * 256 + threadIdx.x;
    if (t < n) p[t] = 0.0f;
}
__global__ void set_i_kernel(int* p, int n, int v) {
    int t = blockIdx.x * 256 + threadIdx.x;
    if (t < n) p[t] = v;
}

// xcur = x (f32 working copy), x16 = f16 copy for WMMA
__global__ void init_nodes_kernel(const float* __restrict__ x,
                                  float* __restrict__ xcur,
                                  _Float16* __restrict__ x16, int nd) {
    int t = blockIdx.x * 256 + threadIdx.x;
    if (t >= nd) return;
    float v = x[t];
    xcur[t] = v;
    x16[t]  = (_Float16)v;
}

// Wt16[n*128+k] = (f16) W[k*128+n]   (transpose so B fragments are contiguous)
__global__ void wt_conv_kernel(const float* __restrict__ W, _Float16* __restrict__ Wt) {
    int t = blockIdx.x * 256 + threadIdx.x;   // 0 .. 16383
    int n = t >> 7, k = t & 127;
    Wt[n * 128 + k] = (_Float16)W[k * 128 + n];
}

// ---------------- edge encoder: ef = relu(attr@w1+b1)@w2+b2 ; also sums for mean ---
__global__ void encoder_kernel(const float* __restrict__ attr,
                               const float* __restrict__ w1, const float* __restrict__ b1,
                               const float* __restrict__ w2, const float* __restrict__ b2,
                               float* __restrict__ ef, float* __restrict__ efsum, int E) {
    __shared__ float sw1[3 * 128];
    __shared__ float sb1[128];
    __shared__ float sw2[128 * 4];
    int t = threadIdx.x;
    for (int i = t; i < 384; i += 256) sw1[i] = w1[i];
    if (t < 128) sb1[t] = b1[t];
    for (int i = t; i < 512; i += 256) sw2[i] = w2[i];
    __syncthreads();

    int e = blockIdx.x * 256 + t;
    float f0 = 0.f, f1 = 0.f, f2 = 0.f, f3 = 0.f;
    if (e < E) {
        float a0 = attr[e * 3 + 0], a1 = attr[e * 3 + 1], a2 = attr[e * 3 + 2];
        #pragma unroll 4
        for (int j = 0; j < 128; ++j) {
            float tv = fmaf(a0, sw1[j], fmaf(a1, sw1[128 + j], fmaf(a2, sw1[256 + j], sb1[j])));
            tv = tv > 0.f ? tv : 0.f;
            f0 = fmaf(tv, sw2[j * 4 + 0], f0);
            f1 = fmaf(tv, sw2[j * 4 + 1], f1);
            f2 = fmaf(tv, sw2[j * 4 + 2], f2);
            f3 = fmaf(tv, sw2[j * 4 + 3], f3);
        }
        f0 += b2[0]; f1 += b2[1]; f2 += b2[2]; f3 += b2[3];
        ef[e * 4 + 0] = f0; ef[e * 4 + 1] = f1;
        ef[e * 4 + 2] = f2; ef[e * 4 + 3] = f3;
    }
    // wave-level reduce before the atomic (wave32)
    for (int o = 16; o > 0; o >>= 1) {
        f0 += __shfl_down(f0, o);
        f1 += __shfl_down(f1, o);
        f2 += __shfl_down(f2, o);
        f3 += __shfl_down(f3, o);
    }
    if ((t & 31) == 0) {
        atomicAdd(&efsum[0], f0); atomicAdd(&efsum[1], f1);
        atomicAdd(&efsum[2], f2); atomicAdd(&efsum[3], f3);
    }
}

// efmean = efsum/E ; M[l][j][h] = sum_c W_edge[l][j][h*32+c] * att_edge[l][h*32+c]
__global__ void finalize_kernel(const float* __restrict__ efsum, float* __restrict__ efmean,
                                const float* __restrict__ W_edge,
                                const float* __restrict__ att_edge,
                                float* __restrict__ M, int E) {
    int t = threadIdx.x;
    if (t < 4) efmean[t] = efsum[t] / (float)E;
    if (t < 48) {
        int l = t / 16, rem = t % 16, j = rem >> 2, hh = rem & 3;
        float s = 0.f;
        for (int c = 0; c < 32; ++c)
            s = fmaf(W_edge[l * 512 + j * 128 + hh * 32 + c],
                     att_edge[l * 128 + hh * 32 + c], s);
        M[t] = s;                       // M[l*16 + j*4 + hh]
    }
}

// write src/dst (with self loops) into output tail as int bit patterns
__global__ void idx_kernel(const int* __restrict__ ei, int* __restrict__ out_idx,
                           int E, int Ef) {
    int i = blockIdx.x * 256 + threadIdx.x;
    if (i >= Ef) return;
    out_idx[i]      = (i < E) ? ei[i]     : (i - E);
    out_idx[Ef + i] = (i < E) ? ei[E + i] : (i - E);
}

// ---------------- h = x16 @ Wt16  via v_wmma_f32_16x16x32_f16 --------------
// block = 256 threads = 8 waves; block computes 16 rows x 128 cols of h.
__global__ void gemm_h_wmma(const _Float16* __restrict__ x16,
                            const _Float16* __restrict__ Wt,
                            float* __restrict__ h) {
    __shared__ __align__(16) _Float16 tileA[16 * 128];
    const int row0 = blockIdx.x * 16;
    const int t    = threadIdx.x;

    // cooperative A-tile load: 256 threads x 8 halves = 16x128
    {
        int r  = t >> 4;
        int c8 = (t & 15) << 3;
        *(v8h*)&tileA[r * 128 + c8] = *(const v8h*)&x16[(row0 + r) * 128 + c8];
    }
    __syncthreads();

    const int lane  = t & 31;
    const int wave  = t >> 5;
    const int l15   = lane & 15;
    const int khalf = (lane < 16) ? 0 : 8;          // K-half per ISA 16-bit layout
    const int ncol  = (wave << 4) + l15;            // B column this lane serves

    v8f c = {};
    #pragma unroll
    for (int k0 = 0; k0 < 128; k0 += 32) {
        Frag a, b;
        a.h[0] = *(const v8h*)&tileA[l15 * 128 + k0 + khalf];
        a.h[1] = *(const v8h*)&tileA[l15 * 128 + k0 + khalf + 16];
        b.h[0] = *(const v8h*)&Wt[ncol * 128 + k0 + khalf];
        b.h[1] = *(const v8h*)&Wt[ncol * 128 + k0 + khalf + 16];
        c = __builtin_amdgcn_wmma_f32_16x16x32_f16(false, a.v, false, b.v,
                                                   (short)0, c, false, false);
    }

    // C/D layout: VGPR r -> M=r (lanes 0-15) / M=r+8 (lanes 16-31); N = lane&15
    const int mbase = row0 + ((lane < 16) ? 0 : 8);
    const int col   = (wave << 4) + l15;
    #pragma unroll
    for (int r = 0; r < 8; ++r)
        h[(mbase + r) * 128 + col] = c[r];
}

// alpha_src / alpha_dst per (node, head)
__global__ void alpha_sd_kernel(const float* __restrict__ h,
                                const float* __restrict__ a_src,
                                const float* __restrict__ a_dst,
                                float* __restrict__ asrc, float* __restrict__ adst, int n) {
    int t = blockIdx.x * 256 + threadIdx.x;
    if (t >= n * 4) return;
    int i = t >> 2, hh = t & 3;
    float s1 = 0.f, s2 = 0.f;
    #pragma unroll 4
    for (int c = 0; c < 32; ++c) {
        float hv = h[i * 128 + hh * 32 + c];
        s1 = fmaf(hv, a_src[hh * 32 + c], s1);
        s2 = fmaf(hv, a_dst[hh * 32 + c], s2);
    }
    asrc[t] = s1; adst[t] = s2;
}

// pass 1: e = leaky(asrc[src]+adst[dst]+ef@M); segment max via encoded atomicMax
__global__ void edgeA_kernel(const int* __restrict__ ei, const float* __restrict__ ef,
                             const float* __restrict__ efmean, const float* __restrict__ Ml,
                             const float* __restrict__ asrc, const float* __restrict__ adst,
                             float* __restrict__ e_ex, int* __restrict__ menc,
                             int E, int Ef) {
    int t = blockIdx.x * 256 + threadIdx.x;
    if (t >= Ef * 4) return;
    int edge = t >> 2, hh = t & 3;
    int src, dst; const float* efp;
    if (edge < E) { src = ei[edge]; dst = ei[E + edge]; efp = &ef[edge * 4]; }
    else          { src = dst = edge - E;               efp = efmean; }
    float ae = fmaf(efp[0], Ml[hh],
               fmaf(efp[1], Ml[4 + hh],
               fmaf(efp[2], Ml[8 + hh], efp[3] * Ml[12 + hh])));
    float ev = asrc[src * 4 + hh] + adst[dst * 4 + hh] + ae;
    ev = ev > 0.f ? ev : NEG_SLOPE * ev;
    e_ex[t] = ev;
    atomicMax(&menc[dst * 4 + hh], enc_f(ev));
}

// pass 2: ex = exp(e - max[dst]); denom[dst] += ex  (ex overwrites e in place)
__global__ void edgeB_kernel(const int* __restrict__ ei, float* __restrict__ e_ex,
                             const int* __restrict__ menc, float* __restrict__ denom,
                             int E, int Ef) {
    int t = blockIdx.x * 256 + threadIdx.x;
    if (t >= Ef * 4) return;
    int edge = t >> 2, hh = t & 3;
    int dst = (edge < E) ? ei[E + edge] : (edge - E);
    float ex = __expf(e_ex[t] - dec_f(menc[dst * 4 + hh]));
    e_ex[t] = ex;
    atomicAdd(&denom[dst * 4 + hh], ex);
}

// pass 3: alpha = ex/denom[dst]; scatter msg; emit attn output for this layer
__global__ void edgeC_kernel(const int* __restrict__ ei, const float* __restrict__ e_ex,
                             const float* __restrict__ denom, const float* __restrict__ hmat,
                             float* __restrict__ acc, float* __restrict__ attn_out,
                             int E, int Ef) {
    int t = blockIdx.x * 256 + threadIdx.x;       // Ef*128 < 2^31
    int edge = t >> 7;
    if (edge >= Ef) return;
    int d = t & 127;
    int src, dst;
    if (edge < E) { src = ei[edge]; dst = ei[E + edge]; }
    else          { src = dst = edge - E; }
    int hh = d >> 5;
    float alpha = e_ex[edge * 4 + hh] / denom[dst * 4 + hh];
    atomicAdd(&acc[dst * 128 + d], hmat[src * 128 + d] * alpha);
    if (d < 4)
        attn_out[edge * 4 + d] = e_ex[edge * 4 + d] / denom[dst * 4 + d];
}

// residual + LayerNorm; refresh f32 and f16 copies of x; optionally emit final x
__global__ void ln_kernel(float* __restrict__ xcur, const float* __restrict__ acc,
                          const float* __restrict__ bias, const float* __restrict__ gamma,
                          const float* __restrict__ beta, _Float16* __restrict__ x16,
                          float* __restrict__ out_x) {
    __shared__ float s1[128];
    __shared__ float s2[128];
    int row = blockIdx.x, t = threadIdx.x;
    float v = xcur[row * 128 + t] + acc[row * 128 + t] + bias[t];
    s1[t] = v; s2[t] = v * v;
    __syncthreads();
    #pragma unroll
    for (int s = 64; s > 0; s >>= 1) {
        if (t < s) { s1[t] += s1[t + s]; s2[t] += s2[t + s]; }
        __syncthreads();
    }
    float mu  = s1[0] * (1.0f / 128.0f);
    float var = s2[0] * (1.0f / 128.0f) - mu * mu;
    float y = (v - mu) * rsqrtf(var + LN_EPS) * gamma[t] + beta[t];
    xcur[row * 128 + t] = y;
    x16[row * 128 + t]  = (_Float16)y;
    if (out_x) out_x[row * 128 + t] = y;
}

// ---------------------------------------------------------------------------
static inline unsigned gr(long long n, int b) { return (unsigned)((n + b - 1) / b); }

extern "C" void kernel_launch(void* const* d_in, const int* in_sizes, int n_in,
                              void* d_out, int out_size, void* d_ws, size_t ws_size,
                              hipStream_t stream) {
    const float* x        = (const float*)d_in[0];
    const int*   ei       = (const int*)  d_in[1];
    const float* attr     = (const float*)d_in[2];
    const float* enc_w1   = (const float*)d_in[3];
    const float* enc_b1   = (const float*)d_in[4];
    const float* enc_w2   = (const float*)d_in[5];
    const float* enc_b2   = (const float*)d_in[6];
    const float* Ws       = (const float*)d_in[7];
    const float* att_src  = (const float*)d_in[8];
    const float* att_dst  = (const float*)d_in[9];
    const float* att_edge = (const float*)d_in[10];
    const float* W_edge   = (const float*)d_in[11];
    const float* biases   = (const float*)d_in[12];
    const float* ln_gamma = (const float*)d_in[13];
    const float* ln_beta  = (const float*)d_in[14];

    const int N  = in_sizes[0] / D_;
    const int E  = in_sizes[1] / 2;
    const int Ef = E + N;
    const long long ND = (long long)N * D_;

    // ---- workspace sub-allocation (256B aligned) ----
    size_t off = 0;
    auto take = [&](size_t bytes) -> char* {
        off = (off + 255) & ~(size_t)255;
        char* p = (char*)d_ws + off;
        off += bytes;
        return p;
    };
    _Float16* x16    = (_Float16*)take(ND * sizeof(_Float16));
    _Float16* Wt16   = (_Float16*)take((size_t)D_ * D_ * sizeof(_Float16));
    float*    xcur   = (float*)take(ND * sizeof(float));
    float*    hmat   = (float*)take(ND * sizeof(float));
    float*    acc    = (float*)take(ND * sizeof(float));
    float*    asrc   = (float*)take((size_t)N * H_ * sizeof(float));
    float*    adst   = (float*)take((size_t)N * H_ * sizeof(float));
    float*    ef     = (float*)take((size_t)E * H_ * sizeof(float));
    float*    e_ex   = (float*)take((size_t)Ef * H_ * sizeof(float));
    int*      menc   = (int*)  take((size_t)N * H_ * sizeof(int));
    float*    denom  = (float*)take((size_t)N * H_ * sizeof(float));
    float*    efsum  = (float*)take(8 * sizeof(float));   // sums[4]
    float*    efmean = (float*)take(8 * sizeof(float));   // mean[4]
    float*    Mmat   = (float*)take(64 * sizeof(float));  // M[l*16+j*4+h], l<3
    (void)ws_size; (void)n_in; (void)out_size;

    // ---- output layout ----
    float* out_x    = (float*)d_out;                      // N*D
    float* out_attn = out_x + ND;                         // L*Ef*H
    int*   out_idx  = (int*)(out_attn + (size_t)L_ * Ef * H_);  // 2*Ef

    // ---- prologue ----
    init_nodes_kernel<<<gr(ND, 256), 256, 0, stream>>>(x, xcur, x16, (int)ND);
    zero_f_kernel<<<1, 256, 0, stream>>>(efsum, 8);
    encoder_kernel<<<gr(E, 256), 256, 0, stream>>>(attr, enc_w1, enc_b1, enc_w2, enc_b2,
                                                   ef, efsum, E);
    finalize_kernel<<<1, 64, 0, stream>>>(efsum, efmean, W_edge, att_edge, Mmat, E);
    idx_kernel<<<gr(Ef, 256), 256, 0, stream>>>(ei, out_idx, E, Ef);

    // ---- layers ----
    for (int l = 0; l < L_; ++l) {
        wt_conv_kernel<<<64, 256, 0, stream>>>(Ws + (size_t)l * D_ * D_, Wt16);
        gemm_h_wmma<<<N / 16, 256, 0, stream>>>(x16, Wt16, hmat);
        alpha_sd_kernel<<<gr((long long)N * H_, 256), 256, 0, stream>>>(
            hmat, att_src + l * D_, att_dst + l * D_, asrc, adst, N);

        zero_f_kernel<<<gr(ND, 256), 256, 0, stream>>>(acc, (int)ND);
        zero_f_kernel<<<gr((long long)N * H_, 256), 256, 0, stream>>>(denom, N * H_);
        set_i_kernel<<<gr((long long)N * H_, 256), 256, 0, stream>>>(menc, N * H_, INT_MIN);

        edgeA_kernel<<<gr((long long)Ef * H_, 256), 256, 0, stream>>>(
            ei, ef, efmean, Mmat + l * 16, asrc, adst, e_ex, menc, E, Ef);
        edgeB_kernel<<<gr((long long)Ef * H_, 256), 256, 0, stream>>>(
            ei, e_ex, menc, denom, E, Ef);
        edgeC_kernel<<<gr((long long)Ef * D_, 256), 256, 0, stream>>>(
            ei, e_ex, denom, hmat, acc, out_attn + (size_t)l * Ef * H_, E, Ef);

        ln_kernel<<<N, 128, 0, stream>>>(xcur, acc, biases + l * D_,
                                         ln_gamma + l * D_, ln_beta + l * D_, x16,
                                         (l == L_ - 1) ? out_x : nullptr);
    }
}
